// AdditiveAttention_49658411876681
// MI455X (gfx1250) — matrix operand order repaired and checked by reading, not compiled, not run on hardware
//
#include <hip/hip_runtime.h>

// B=32, S=2048, E=1024, D=1024, A=1024
#define BB 32
#define SS 2048
#define EE 1024
#define DD 1024
#define AA 1024

typedef __attribute__((ext_vector_type(16))) __bf16 v16bf;
typedef __attribute__((ext_vector_type(8)))  float  v8f;

union Frag16 { v16bf v; uint4 q[2]; };

__device__ __forceinline__ unsigned f32_to_bf16_rne(float f) {
    unsigned u = __float_as_uint(f);
    return (u + 0x7FFFu + ((u >> 16) & 1u)) >> 16;
}

// Branch-free tanh on the native v_exp_f32 / v_rcp_f32 path:
// tanh(x) = 1 - 2/(exp(2x)+1); saturates correctly at +/-inf.
__device__ __forceinline__ float tanh_fast(float x) {
    float e = __expf(2.0f * x);
    return 1.0f - 2.0f * __builtin_amdgcn_rcpf(e + 1.0f);
}

// ---- Kernel 1: W_enc f32 -> bf16 (4 MB read, 2 MB write, once per launch)
__global__ __launch_bounds__(256) void wenc_to_bf16_kernel(const float* __restrict__ W,
                                                           unsigned short* __restrict__ Wb) {
    int i = (blockIdx.x * 256 + threadIdx.x) * 4;
    float4 f = *(const float4*)(W + i);
    unsigned lo = f32_to_bf16_rne(f.x) | (f32_to_bf16_rne(f.y) << 16);
    unsigned hi = f32_to_bf16_rne(f.z) | (f32_to_bf16_rne(f.w) << 16);
    uint2 p; p.x = lo; p.y = hi;
    *(uint2*)(Wb + i) = p;
}

// ---- Kernel 2: dec_proj[b][a] = sum_d dec[b][d] * W_dec[a][d]  (one wave per output)
__global__ __launch_bounds__(256) void decproj_kernel(const float* __restrict__ dec,
                                                      const float* __restrict__ Wd,
                                                      float* __restrict__ dp) {
    int wave = threadIdx.x >> 5, lane = threadIdx.x & 31;
    int o = blockIdx.x * 8 + wave;            // 0 .. B*A-1
    int b = o >> 10, a = o & (AA - 1);
    const float* dr = dec + (size_t)b * DD;
    const float* wr = Wd + (size_t)a * DD;
    float acc = 0.f;
    for (int k = lane; k < DD; k += 32) acc += dr[k] * wr[k];
    for (int m = 16; m > 0; m >>= 1) acc += __shfl_xor(acc, m, 32);
    if (lane == 0) dp[o] = acc;
}

// ---- Kernel 3: fused scores.
// grid (S/32, B), 256 threads (8 waves). 32-row M-tile staged as bf16 in LDS.
// Each wave: 2 m-frags x 2 a-tiles = 4 accumulators; B-frags reused across m-frags
// (halves W_enc L2 traffic vs a 16-row tile), A-frags reused across a-tiles.
#define LPAD 1032   // 1024+8 halves pad -> row stride 516 dwords, conflict-free rows
__global__ __launch_bounds__(256) void scores_kernel(const float* __restrict__ enc,
                                                     const unsigned short* __restrict__ Wb,
                                                     const float* __restrict__ decp,
                                                     const float* __restrict__ vvec,
                                                     float* __restrict__ scores) {
    __shared__ unsigned short encT[32 * LPAD];
    __shared__ float sLDS[32];

    const int b   = blockIdx.y;
    const int s0  = blockIdx.x * 32;
    const int tid = threadIdx.x;
    if (tid < 32) sLDS[tid] = 0.f;

    // Stage 32 x 1024 encoder rows as bf16 into LDS (coalesced float2 loads).
    const float* encBase = enc + ((size_t)b * SS + s0) * EE;
    #pragma unroll 4
    for (int j = 0; j < 64; ++j) {
        int p   = j * 256 + tid;          // pair index, 16384 pairs
        int row = p >> 9;                 // 512 pairs per row
        int c2  = (p & 511) << 1;
        float2 f = *(const float2*)(encBase + (size_t)row * EE + c2);
        unsigned pk = f32_to_bf16_rne(f.x) | (f32_to_bf16_rne(f.y) << 16);
        *(unsigned*)&encT[row * LPAD + c2] = pk;
    }
    __syncthreads();

    const int lane = tid & 31, wave = tid >> 5;
    const int lw = lane & 15, h = lane >> 4;

    // A-fragment bases (ISA 16-bit A layout): lane = row, K chunks at h*8 and h*8+16.
    const unsigned short* ap0 = encT + lw * LPAD + h * 8;          // rows 0..15
    const unsigned short* ap1 = encT + (16 + lw) * LPAD + h * 8;   // rows 16..31

    float sacc0[8], sacc1[8];
    #pragma unroll
    for (int r = 0; r < 8; ++r) { sacc0[r] = 0.f; sacc1[r] = 0.f; }

    #pragma unroll 1
    for (int t = 0; t < 4; ++t) {
        int a0 = ((t * 8 + wave) * 2) * 16;   // two adjacent 16-wide a-tiles per wave
        int a1 = a0 + 16;
        v8f c00 = {}, c01 = {}, c10 = {}, c11 = {};
        // B-fragment (ISA 16-bit B layout): lane = column, K chunks at h*8 and h*8+16.
        const unsigned short* wb0 = Wb + (size_t)(a0 + lw) * EE + h * 8;
        const unsigned short* wb1 = Wb + (size_t)(a1 + lw) * EE + h * 8;

        #pragma unroll 2
        for (int k = 0; k < EE; k += 32) {
            Frag16 af0, af1, bf0, bf1;
            af0.q[0] = *(const uint4*)(ap0 + k);
            af0.q[1] = *(const uint4*)(ap0 + k + 16);
            af1.q[0] = *(const uint4*)(ap1 + k);
            af1.q[1] = *(const uint4*)(ap1 + k + 16);
            bf0.q[0] = *(const uint4*)(wb0 + k);
            bf0.q[1] = *(const uint4*)(wb0 + k + 16);
            bf1.q[0] = *(const uint4*)(wb1 + k);
            bf1.q[1] = *(const uint4*)(wb1 + k + 16);
            c00 = __builtin_amdgcn_wmma_f32_16x16x32_bf16(false, af0.v, false, bf0.v,
                                                          (short)0, c00, false, false);
            c01 = __builtin_amdgcn_wmma_f32_16x16x32_bf16(false, af0.v, false, bf1.v,
                                                          (short)0, c01, false, false);
            c10 = __builtin_amdgcn_wmma_f32_16x16x32_bf16(false, af1.v, false, bf0.v,
                                                          (short)0, c10, false, false);
            c11 = __builtin_amdgcn_wmma_f32_16x16x32_bf16(false, af1.v, false, bf1.v,
                                                          (short)0, c11, false, false);
        }
        // epilogue: + dec_proj, tanh, * v, per-lane partial a-reduction
        float dp0 = decp[b * AA + a0 + lw], vv0 = vvec[a0 + lw];
        float dp1 = decp[b * AA + a1 + lw], vv1 = vvec[a1 + lw];
        #pragma unroll
        for (int r = 0; r < 8; ++r) {
            sacc0[r] += tanh_fast(c00[r] + dp0) * vv0 + tanh_fast(c01[r] + dp1) * vv1;
            sacc1[r] += tanh_fast(c10[r] + dp0) * vv0 + tanh_fast(c11[r] + dp1) * vv1;
        }
    }

    // reduce the 16 columns owned by each half-wave (masks <=8 stay inside the half)
    #pragma unroll
    for (int m = 1; m <= 8; m <<= 1) {
        #pragma unroll
        for (int r = 0; r < 8; ++r) {
            sacc0[r] += __shfl_xor(sacc0[r], m, 32);
            sacc1[r] += __shfl_xor(sacc1[r], m, 32);
        }
    }
    // C layout: VGPR r, half h -> row r + 8*h (m-frag0) and 16 + r + 8*h (m-frag1).
    #pragma unroll
    for (int r = 0; r < 8; ++r) {
        if (lw == r) {
            atomicAdd(&sLDS[h * 8 + r],      sacc0[r]);
            atomicAdd(&sLDS[16 + h * 8 + r], sacc1[r]);
        }
    }
    __syncthreads();
    if (tid < 32) scores[(size_t)b * SS + s0 + tid] = sLDS[tid];
}

// ---- Kernel 4: softmax over S per batch row
__global__ __launch_bounds__(256) void softmax_kernel(const float* __restrict__ scores,
                                                      float* __restrict__ attn) {
    __shared__ float red[256];
    int b = blockIdx.x, tid = threadIdx.x;
    const float* row = scores + (size_t)b * SS;
    float m = -INFINITY;
    for (int s = tid; s < SS; s += 256) m = fmaxf(m, row[s]);
    red[tid] = m; __syncthreads();
    for (int o = 128; o > 0; o >>= 1) { if (tid < o) red[tid] = fmaxf(red[tid], red[tid + o]); __syncthreads(); }
    m = red[0]; __syncthreads();
    float sum = 0.f;
    for (int s = tid; s < SS; s += 256) sum += __expf(row[s] - m);
    red[tid] = sum; __syncthreads();
    for (int o = 128; o > 0; o >>= 1) { if (tid < o) red[tid] += red[tid + o]; __syncthreads(); }
    float inv = 1.f / red[0];
    for (int s = tid; s < SS; s += 256) attn[(size_t)b * SS + s] = __expf(row[s] - m) * inv;
}

// ---- Kernel 5: zero context accumulator (d_out is poisoned)
__global__ __launch_bounds__(256) void ctx_zero_kernel(float* __restrict__ ctx) {
    ctx[blockIdx.x * 256 + threadIdx.x] = 0.f;
}

// ---- Kernel 6: context[b][e] += sum_s attn[b][s] * enc[b][s][e]  (split-S, f32 atomics)
__global__ __launch_bounds__(256) void context_kernel(const float* __restrict__ enc,
                                                      const float* __restrict__ attn,
                                                      float* __restrict__ ctx) {
    __shared__ float w[512];
    int b = blockIdx.z;
    int e = blockIdx.x * 256 + threadIdx.x;
    int sBase = blockIdx.y * 512;
    w[threadIdx.x]       = attn[(size_t)b * SS + sBase + threadIdx.x];
    w[threadIdx.x + 256] = attn[(size_t)b * SS + sBase + 256 + threadIdx.x];
    __syncthreads();
    float acc = 0.f;
    const float* p = enc + ((size_t)b * SS + sBase) * EE + e;
    #pragma unroll 4
    for (int s = 0; s < 512; ++s) acc += w[s] * p[(size_t)s * EE];
    atomicAdd(&ctx[(size_t)b * EE + e], acc);
}

extern "C" void kernel_launch(void* const* d_in, const int* in_sizes, int n_in,
                              void* d_out, int out_size, void* d_ws, size_t ws_size,
                              hipStream_t stream) {
    const float* enc   = (const float*)d_in[0];   // (B,S,E)
    const float* dec   = (const float*)d_in[1];   // (B,D)
    const float* W_enc = (const float*)d_in[2];   // (A,E)
    const float* W_dec = (const float*)d_in[3];   // (A,D)
    const float* vvec  = (const float*)d_in[4];   // (A)

    float* ctx  = (float*)d_out;                  // (B,E)
    float* attn = (float*)d_out + BB * EE;        // (B,S)

    char* ws = (char*)d_ws;
    unsigned short* Wb = (unsigned short*)ws;                       // 2 MB bf16 W_enc
    float* dp     = (float*)(ws + (size_t)AA * EE * 2);             // B*A f32
    float* scores = (float*)(ws + (size_t)AA * EE * 2 + (size_t)BB * AA * 4); // B*S f32

    wenc_to_bf16_kernel<<<AA * EE / (256 * 4), 256, 0, stream>>>(W_enc, Wb);
    decproj_kernel<<<BB * AA / 8, 256, 0, stream>>>(dec, W_dec, dp);
    scores_kernel<<<dim3(SS / 32, BB), 256, 0, stream>>>(enc, Wb, dp, vvec, scores);
    softmax_kernel<<<BB, 256, 0, stream>>>(scores, attn);
    ctx_zero_kernel<<<BB * EE / 256, 256, 0, stream>>>(ctx);
    context_kernel<<<dim3(EE / 256, SS / 512, BB), 256, 0, stream>>>(enc, attn, ctx);
}
